// GCN_17428977287558
// MI455X (gfx1250) — matrix operand order repaired and checked by reading, not compiled, not run on hardware
//
#include <hip/hip_runtime.h>
#include <hip/hip_bf16.h>

// ---------------------------------------------------------------------------
// GCN (3x GraphConv, norm='both') for MI455X / gfx1250.
//   per layer: hs = h * norm_src ; agg = scatter_add(hs[src] -> dst)
//              a  = bf16(agg * norm_dst) ; out = relu?(a @ W + b)  via WMMA bf16
// ---------------------------------------------------------------------------

typedef __attribute__((ext_vector_type(16))) __bf16 v16bf;
typedef __attribute__((ext_vector_type(8)))  float  v8f;

union FragAB {
    uint4 u[2];   // 32 bytes = 16 bf16
    v16bf v;
};

static __device__ __forceinline__ unsigned short f2bf(float f) {
    unsigned int u = __float_as_uint(f);
    unsigned int r = (u + 0x7FFFu + ((u >> 16) & 1u)) >> 16;   // round-to-nearest-even
    return (unsigned short)r;
}

// ---------------------------- small utility kernels ------------------------

__global__ void zero_f4_kernel(float4* __restrict__ p, int n4) {
    int i = blockIdx.x * blockDim.x + threadIdx.x;
    if (i < n4) p[i] = make_float4(0.f, 0.f, 0.f, 0.f);
}

// deg accumulation: one thread per edge, float atomics (counts << 2^24)
__global__ void degree_kernel(const long long* __restrict__ src,
                              const long long* __restrict__ dst,
                              float* __restrict__ deg_out,
                              float* __restrict__ deg_in, int E) {
    int e = blockIdx.x * blockDim.x + threadIdx.x;
    if (e >= E) return;
    unsafeAtomicAdd(deg_out + (int)src[e], 1.0f);
    unsafeAtomicAdd(deg_in  + (int)dst[e], 1.0f);
}

// norm = rsqrt(max(deg,1)) in place; operates on 2N contiguous floats
__global__ void finalize_norm_kernel(float* __restrict__ deg, int n) {
    int i = blockIdx.x * blockDim.x + threadIdx.x;
    if (i < n) deg[i] = rsqrtf(fmaxf(deg[i], 1.0f));
}

// Wt[n*K + k] = bf16(W[k*Nout + n]), zero-padded to NoutPad rows
__global__ void convert_weight_kernel(const float* __restrict__ W,
                                      unsigned short* __restrict__ Wt,
                                      int K, int Nout, int NoutPad) {
    int i = blockIdx.x * blockDim.x + threadIdx.x;
    if (i >= NoutPad * K) return;
    int n = i / K, k = i - n * K;
    float v = (n < Nout) ? W[k * Nout + n] : 0.f;
    Wt[i] = f2bf(v);
}

// hs[i,:] = h[i,:] * norm[i]   (float4 per thread, 32 threads per row of 128)
__global__ void scale_rows_kernel(const float* __restrict__ h,
                                  const float* __restrict__ norm,
                                  float* __restrict__ hs, int n4) {
    int i = blockIdx.x * blockDim.x + threadIdx.x;
    if (i >= n4) return;
    int row = i >> 5;
    float s = norm[row];
    float4 v = reinterpret_cast<const float4*>(h)[i];
    v.x *= s; v.y *= s; v.z *= s; v.w *= s;
    reinterpret_cast<float4*>(hs)[i] = v;
}

// abf[i,:] = bf16(agg[i,:] * norm_dst[i])
__global__ void scale_to_bf16_kernel(const float* __restrict__ agg,
                                     const float* __restrict__ norm,
                                     unsigned short* __restrict__ abf, int n4) {
    int i = blockIdx.x * blockDim.x + threadIdx.x;
    if (i >= n4) return;
    int row = i >> 5;
    float s = norm[row];
    float4 v = reinterpret_cast<const float4*>(agg)[i];
    ushort4 o;
    o.x = f2bf(v.x * s); o.y = f2bf(v.y * s);
    o.z = f2bf(v.z * s); o.w = f2bf(v.w * s);
    reinterpret_cast<ushort4*>(abf)[i] = o;
}

// one wave32 per edge: lane l moves float4 at feature offset 4*l
// global_load_b128 from hs[src] + 4x global_atomic_add_f32 into agg[dst]
__global__ void edge_scatter_kernel(const float* __restrict__ hs,
                                    const long long* __restrict__ src,
                                    const long long* __restrict__ dst,
                                    float* __restrict__ agg, int E) {
    int wid  = (blockIdx.x * blockDim.x + threadIdx.x) >> 5;
    int lane = threadIdx.x & 31;
    if (wid >= E) return;
    long long s = src[wid];
    long long d = dst[wid];
    const float4 v = *reinterpret_cast<const float4*>(hs + s * 128 + lane * 4);
    float* out = agg + d * 128 + lane * 4;
    unsafeAtomicAdd(out + 0, v.x);
    unsafeAtomicAdd(out + 1, v.y);
    unsafeAtomicAdd(out + 2, v.z);
    unsafeAtomicAdd(out + 3, v.w);
}

// ------------------------------- WMMA GEMM ---------------------------------
// C[M x Nout] = A[M x 128 bf16] @ W[128 x Nout] + bias ; optional ReLU.
// Wt is the bf16 weight pre-transposed to [NTILES*16 x 128].
// One wave per 16-row strip of C; all 4 K-fragments of A held in registers;
// bias preloaded into registers; all 4 B fragments loaded before the WMMA
// chain so the backend can issue a single 8x b128 clause per N-tile.
template <bool RELU, int NTILES>
__global__ __launch_bounds__(256)
void gemm_wmma_kernel(const unsigned short* __restrict__ A,
                      const unsigned short* __restrict__ Wt,
                      const float* __restrict__ bias,
                      float* __restrict__ C,
                      int M, int Nout, int ldc) {
    const int lane  = threadIdx.x & 31;
    const int wave  = threadIdx.x >> 5;
    const int strip = blockIdx.x * (blockDim.x >> 5) + wave;
    if (strip * 16 >= M) return;            // wave-uniform: EXEC stays all-ones

    const int m0      = strip * 16;
    const int halfsel = lane >> 4;          // 0: lanes 0-15, 1: lanes 16-31
    const int r       = lane & 15;

    // Preload bias for every N-tile this wave touches (out of the hot loop).
    float bv[NTILES];
#pragma unroll
    for (int nt = 0; nt < NTILES; ++nt) {
        const int col = nt * 16 + r;
        bv[nt] = (col < Nout) ? bias[col] : 0.f;
    }

    // A fragment per ISA 16-bit A 16x32 layout:
    //   row = m0 + (lane&15); elems 0-7 -> K = k0+off+{0..7}; elems 8-15 -> +16
    FragAB af[4];
    {
        const int arow = m0 + r;
        const int aoff = halfsel * 8;
        const unsigned short* abase = A + (size_t)arow * 128 + aoff;
#pragma unroll
        for (int kt = 0; kt < 4; ++kt) {
            const uint4* pa = reinterpret_cast<const uint4*>(abase + kt * 32);
            af[kt].u[0] = pa[0];            // K = k0+off+0..7
            af[kt].u[1] = pa[2];            // K = k0+off+16..23
        }
    }

    const int boff = halfsel * 8;           // B: same K order as A (off+0..7, off+16..23)
#pragma unroll
    for (int nt = 0; nt < NTILES; ++nt) {
        const int col = nt * 16 + r;
        const unsigned short* bbase = Wt + (size_t)col * 128 + boff;

        // Load all 4 B fragments first -> one wide load clause, overlapped
        // against the WMMA chain below.
        FragAB bfr[4];
#pragma unroll
        for (int kt = 0; kt < 4; ++kt) {
            const uint4* pb = reinterpret_cast<const uint4*>(bbase + kt * 32);
            bfr[kt].u[0] = pb[0];
            bfr[kt].u[1] = pb[2];
        }

        const float b = bv[nt];
        v8f acc = {b, b, b, b, b, b, b, b};
#pragma unroll
        for (int kt = 0; kt < 4; ++kt) {
            acc = __builtin_amdgcn_wmma_f32_16x16x32_bf16(
                false, af[kt].v, false, bfr[kt].v, (short)0, acc, false, false);
        }

        if (col < Nout) {                   // divergence only on the store
#pragma unroll
            for (int v = 0; v < 8; ++v) {
                int row = m0 + v + halfsel * 8;
                float o = acc[v];
                if (RELU) o = fmaxf(o, 0.f);
                C[(size_t)row * ldc + col] = o;
            }
        }
    }
}

// ------------------------------ host driver --------------------------------

static inline int cdiv(int a, int b) { return (a + b - 1) / b; }

extern "C" void kernel_launch(void* const* d_in, const int* in_sizes, int n_in,
                              void* d_out, int out_size, void* d_ws, size_t ws_size,
                              hipStream_t stream) {
    const int D = 128;                       // D_IN == D_HID == 128
    const int N = in_sizes[0] / D;           // 50000
    const int E = in_sizes[1] / 2;           // 800000
    const int NC = in_sizes[7];              // 40 classes
    const int NCP = 48;                      // 40 padded to 3 x 16

    const float*     features = (const float*)d_in[0];
    const long long* eidx     = (const long long*)d_in[1];
    const long long* src      = eidx;
    const long long* dst      = eidx + E;
    const float* W1 = (const float*)d_in[2];
    const float* b1 = (const float*)d_in[3];
    const float* W2 = (const float*)d_in[4];
    const float* b2 = (const float*)d_in[5];
    const float* W3 = (const float*)d_in[6];
    const float* b3 = (const float*)d_in[7];
    float* out = (float*)d_out;

    // -------- workspace carve (all chunks 16B aligned) --------
    char* ws = (char*)d_ws;
    float* norm_src = (float*)ws;            ws += (size_t)N * sizeof(float);
    float* norm_dst = (float*)ws;            ws += (size_t)N * sizeof(float);
    float* hs       = (float*)ws;            ws += (size_t)N * D * sizeof(float);
    float* agg      = (float*)ws;            ws += (size_t)N * D * sizeof(float);
    float* h        = (float*)ws;            ws += (size_t)N * D * sizeof(float);
    unsigned short* abf = (unsigned short*)ws; ws += (size_t)N * D * sizeof(unsigned short);
    unsigned short* wt1 = (unsigned short*)ws; ws += (size_t)D * D * sizeof(unsigned short);
    unsigned short* wt2 = (unsigned short*)ws; ws += (size_t)D * D * sizeof(unsigned short);
    unsigned short* wt3 = (unsigned short*)ws; ws += (size_t)NCP * D * sizeof(unsigned short);

    const int TB = 256;
    const int n4      = N * (D / 4);                 // float4 count of an NxD buffer
    const int gScale  = cdiv(n4, TB);
    const int gZero   = cdiv(n4, TB);
    const int gEdgeT  = cdiv(E, TB);                 // 1 thread / edge
    const int gEdgeW  = cdiv(E * 32, TB);            // 1 wave / edge
    const int gGemm   = cdiv(cdiv(N, 16), TB / 32);  // 1 wave / 16-row strip

    // -------- degrees -> norms (recomputed every call: deterministic) --------
    zero_f4_kernel<<<cdiv(2 * N / 4, TB), TB, 0, stream>>>((float4*)norm_src, 2 * N / 4);
    degree_kernel<<<gEdgeT, TB, 0, stream>>>(src, dst, norm_src, norm_dst, E);
    finalize_norm_kernel<<<cdiv(2 * N, TB), TB, 0, stream>>>(norm_src, 2 * N);

    // -------- weights -> bf16, transposed [NoutPad x K] --------
    convert_weight_kernel<<<cdiv(D * D, TB), TB, 0, stream>>>(W1, wt1, D, D, D);
    convert_weight_kernel<<<cdiv(D * D, TB), TB, 0, stream>>>(W2, wt2, D, D, D);
    convert_weight_kernel<<<cdiv(NCP * D, TB), TB, 0, stream>>>(W3, wt3, D, NC, NCP);

    // -------- layer 1: features -> h --------
    scale_rows_kernel<<<gScale, TB, 0, stream>>>(features, norm_src, hs, n4);
    zero_f4_kernel<<<gZero, TB, 0, stream>>>((float4*)agg, n4);
    edge_scatter_kernel<<<gEdgeW, TB, 0, stream>>>(hs, src, dst, agg, E);
    scale_to_bf16_kernel<<<gScale, TB, 0, stream>>>(agg, norm_dst, abf, n4);
    gemm_wmma_kernel<true, 8><<<gGemm, TB, 0, stream>>>(abf, wt1, b1, h, N, D, D);

    // -------- layer 2: h -> h --------
    scale_rows_kernel<<<gScale, TB, 0, stream>>>(h, norm_src, hs, n4);
    zero_f4_kernel<<<gZero, TB, 0, stream>>>((float4*)agg, n4);
    edge_scatter_kernel<<<gEdgeW, TB, 0, stream>>>(hs, src, dst, agg, E);
    scale_to_bf16_kernel<<<gScale, TB, 0, stream>>>(agg, norm_dst, abf, n4);
    gemm_wmma_kernel<true, 8><<<gGemm, TB, 0, stream>>>(abf, wt2, b2, h, N, D, D);

    // -------- layer 3: h -> out [N x NC] --------
    scale_rows_kernel<<<gScale, TB, 0, stream>>>(h, norm_src, hs, n4);
    zero_f4_kernel<<<gZero, TB, 0, stream>>>((float4*)agg, n4);
    edge_scatter_kernel<<<gEdgeW, TB, 0, stream>>>(hs, src, dst, agg, E);
    scale_to_bf16_kernel<<<gScale, TB, 0, stream>>>(agg, norm_dst, abf, n4);
    gemm_wmma_kernel<false, 3><<<gGemm, TB, 0, stream>>>(abf, wt3, b3, out, N, NC, NC);

    (void)n_in; (void)out_size; (void)ws_size;
}
